// EGNN_781684048208
// MI455X (gfx1250) — matrix operand order repaired and checked by reading, not compiled, not run on hardware
//
#include <hip/hip_runtime.h>

// EGNN forward for MI455X (gfx1250, wave32, WMMA).
// All GEMMs run on v_wmma_f32_16x16x32_bf16 (bf16 inputs, fp32 accumulate).
// Node tables (~26MB each) are L2-resident (192MB L2): gathers + segment-sum
// atomics are L2 traffic; edge MLP chain is fully fused per 32-edge tile with
// two WMMAs issued per B-fragment load (M=32 per block, 2 accumulators/wave).
//
// Workspace layout (bytes):
//   [0,           25.6e6)  hcur      : current node features  [N,128] f32
//   [25.6e6,      51.2e6)  agg_feat  : per-layer feature aggregation [N,128] f32
//   [51.2e6,      51.8e6)  agg_coord : per-layer coord aggregation   [N,3]  f32
//   [51.8e6,      52.4e6)  coords_c  : current coords [N,3] f32
//   [52.4e6,      52.6e6)  counts    : incoming-edge counts [N] f32
//   [52.6e6,      53.6e6)  packed bf16 weight fragments (WMMA B layout)
// Total ~53.6 MB.

#define N_NODES 50000
#define N_EDGES 800000
#define IN_CH   64
#define HDIM    128
#define NLAYERS 4

typedef __attribute__((ext_vector_type(16))) __bf16 v16bf;
typedef __attribute__((ext_vector_type(8)))  float  v8f;

__device__ __forceinline__ void atomic_add_f32(float* p, float v) {
  __hip_atomic_fetch_add(p, v, __ATOMIC_RELAXED, __HIP_MEMORY_SCOPE_AGENT);
}

__device__ __forceinline__ float silu_f(float x) {
  return x / (1.f + __expf(-x));
}

// ---------------------------------------------------------------------------
// WMMA tile engines.
// A layout (ISA 7.12.2, 16-bit A 16x32): lanes 0-15: row=lane, halves[0..7]=K
// (kt*32+0..7), halves[8..15]=K(+16..23); lanes 16-31: row=lane-16,
// halves[0..7]=K(+8..15), halves[8..15]=K(+24..31).
// B packed so each lane's 16 halves are contiguous: offset ((kt*8+nt)*32+lane)*16.
// ---------------------------------------------------------------------------
__device__ __forceinline__ v16bf load_a_frag(const __bf16* rowp, int koff) {
  v16bf a;
#pragma unroll
  for (int i = 0; i < 8; ++i) a[i] = rowp[koff + i];
#pragma unroll
  for (int i = 0; i < 8; ++i) a[8 + i] = rowp[koff + 16 + i];
  return a;
}

// Single 16-row M tile.
__device__ __forceinline__ v8f gemm_tile(const __bf16* x, int xstride,
                                         const __bf16* wpk, int KT,
                                         int wave, int lane, v8f acc) {
  const __bf16* rowp = x + (lane & 15) * xstride;
  const int kbase = (lane >= 16) ? 8 : 0;
  for (int kt = 0; kt < KT; ++kt) {
    v16bf a = load_a_frag(rowp, kt * 32 + kbase);
    v16bf b = *(const v16bf*)(wpk + (((kt * 8) + wave) * 32 + lane) * 16);
    acc = __builtin_amdgcn_wmma_f32_16x16x32_bf16(false, a, false, b,
                                                  (short)0, acc, false, false);
  }
  return acc;
}

// Two stacked 16-row M tiles sharing each B fragment (2 WMMAs per B load).
__device__ __forceinline__ void gemm_tile2(const __bf16* x, int xstride,
                                           const __bf16* wpk, int KT,
                                           int wave, int lane,
                                           v8f& acc0, v8f& acc1) {
  const __bf16* rowp0 = x + (lane & 15) * xstride;
  const __bf16* rowp1 = rowp0 + 16 * xstride;
  const int kbase = (lane >= 16) ? 8 : 0;
  for (int kt = 0; kt < KT; ++kt) {
    v16bf a0 = load_a_frag(rowp0, kt * 32 + kbase);
    v16bf a1 = load_a_frag(rowp1, kt * 32 + kbase);
    v16bf b = *(const v16bf*)(wpk + (((kt * 8) + wave) * 32 + lane) * 16);
    acc0 = __builtin_amdgcn_wmma_f32_16x16x32_bf16(false, a0, false, b,
                                                   (short)0, acc0, false, false);
    acc1 = __builtin_amdgcn_wmma_f32_16x16x32_bf16(false, a1, false, b,
                                                   (short)0, acc1, false, false);
  }
}

// ---------------------------------------------------------------------------
// Weight packing: fp32 [K,128] row-major -> bf16 WMMA B-fragment order.
// B layout: for a 32x16 k-tile, lanes 0-15 hold K=0..7,16..23 of column `lane`,
// lanes 16-31 hold K=8..15,24..31 of column lane-16.
// dst flat index = (((kt*8 + nt)*32 + lane)*16 + i).
// ---------------------------------------------------------------------------
__global__ __launch_bounds__(256) void pack_weight(const float* __restrict__ src,
                                                   __bf16* __restrict__ dst,
                                                   int K, int total) {
  int idx = blockIdx.x * 256 + threadIdx.x;
  if (idx >= total) return;
  int i    = idx & 15;
  int lane = (idx >> 4) & 31;
  int nt   = (idx >> 9) & 7;
  int kt   = idx >> 12;
  int col  = nt * 16 + (lane & 15);
  int k    = kt * 32 + ((lane >= 16) ? 8 : 0) + (i & 7) + ((i >= 8) ? 16 : 0);
  dst[idx] = (k < K) ? (__bf16)src[k * HDIM + col] : (__bf16)0.f;
}

__global__ void fill_zero(float* __restrict__ p, int n) {
  int i = blockIdx.x * blockDim.x + threadIdx.x;
  if (i < n) p[i] = 0.f;
}

__global__ void count_kernel(const int* __restrict__ ei, float* __restrict__ counts) {
  int e = blockIdx.x * blockDim.x + threadIdx.x;
  if (e < N_EDGES) atomic_add_f32(&counts[ei[N_EDGES + e]], 1.f);
}

// ---------------------------------------------------------------------------
// Embed-in: hcur = h_in[N,64] @ W[64,128] + b
// ---------------------------------------------------------------------------
__global__ __launch_bounds__(256) void embed_kernel(const float* __restrict__ hin,
                                                    const __bf16* __restrict__ pW,
                                                    const float* __restrict__ bias,
                                                    float* __restrict__ hcur) {
  __shared__ __bf16 xt[16][64];
  const int t = threadIdx.x, wave = t >> 5, lane = t & 31;
  const int n0 = blockIdx.x * 16;
  {
    const int s = t >> 4, j = t & 15, nd = n0 + s;
#pragma unroll
    for (int i = 0; i < 4; ++i) xt[s][j * 4 + i] = (__bf16)hin[nd * IN_CH + j * 4 + i];
  }
  __syncthreads();
  const int cidx = wave * 16 + (lane & 15);
  const int rbase = (lane >= 16) ? 8 : 0;
  v8f acc = {};
  acc = gemm_tile(&xt[0][0], 64, pW, 2, wave, lane, acc);
  const float bv = bias[cidx];
#pragma unroll
  for (int v = 0; v < 8; ++v) hcur[(n0 + rbase + v) * HDIM + cidx] = acc[v] + bv;
}

// ---------------------------------------------------------------------------
// Edge kernel: one block = 32 edges x 128 outputs (8 waves; each wave owns a
// 16-col slice and two stacked 16-edge row tiles -> 2 WMMAs per B fragment).
// Fused: build edge_in -> GEMM1 -> SiLU -> GEMM2 -> SiLU -> atomic agg_feat
// -> GEMM3 -> SiLU -> dot(cWc) -> atomic agg_coord.
// ---------------------------------------------------------------------------
__global__ __launch_bounds__(256) void edge_kernel(
    const int*   __restrict__ ei,      // [2,E]: row (src), then col (dst)
    const float* __restrict__ coords,  // [N,3]
    const float* __restrict__ hcur,    // [N,128]
    const float* __restrict__ eattr,   // [E,1]
    const __bf16* __restrict__ peW1,   // packed [288,128]
    const __bf16* __restrict__ peW2,   // packed [128,128]
    const __bf16* __restrict__ pcW1,   // packed [128,128]
    const float* __restrict__ eb1, const float* __restrict__ eb2,
    const float* __restrict__ cb1, const float* __restrict__ cWc, // [128]
    float* __restrict__ agg_feat, float* __restrict__ agg_coord) {
  __shared__ __bf16 xt[32][288];   // edge_in tile (K padded 258->288 with zeros)
  __shared__ __bf16 t1[32][128];   // after GEMM1+SiLU
  __shared__ __bf16 t2[32][128];   // ef (after GEMM2+SiLU)
  __shared__ float  cf[32][128];   // after GEMM3+SiLU (fp32 for the cWc dot)
  __shared__ int    ncol[32];
  __shared__ float  cdiff[32][3];

  const int t = threadIdx.x;
  const int wave = t >> 5, lane = t & 31;
  const int e0 = blockIdx.x * 32;   // E % 32 == 0

  // ---- build input tile: 8 threads per edge, 16 channels each ----
  {
    const int e = t >> 3;       // edge slot 0..31
    const int j = t & 7;        // channel chunk (16 wide)
    const int ge = e0 + e;
    const int r = ei[ge];               // source j
    const int c = ei[N_EDGES + ge];     // target i
    const float* hc = hcur + c * HDIM + j * 16;
    const float* hr = hcur + r * HDIM + j * 16;
#pragma unroll
    for (int i = 0; i < 16; ++i) xt[e][j * 16 + i] = (__bf16)hc[i];
#pragma unroll
    for (int i = 0; i < 16; ++i) xt[e][HDIM + j * 16 + i] = (__bf16)hr[i];
    if (j == 0) {
      ncol[e] = c;
      const float dx = coords[r * 3 + 0] - coords[c * 3 + 0];
      const float dy = coords[r * 3 + 1] - coords[c * 3 + 1];
      const float dz = coords[r * 3 + 2] - coords[c * 3 + 2];
      cdiff[e][0] = dx; cdiff[e][1] = dy; cdiff[e][2] = dz;
      xt[e][256] = (__bf16)(dx * dx + dy * dy + dz * dz);
      xt[e][257] = (__bf16)eattr[ge];
      for (int p = 258; p < 288; ++p) xt[e][p] = (__bf16)0.f;
    }
  }
  __syncthreads();

  const int cidx = wave * 16 + (lane & 15);
  const int rbase = (lane >= 16) ? 8 : 0;

  // ---- GEMM1: [32,288] @ [288,128], SiLU ----
  {
    v8f a0 = {}, a1 = {};
    gemm_tile2(&xt[0][0], 288, peW1, 9, wave, lane, a0, a1);
    const float bv = eb1[cidx];
#pragma unroll
    for (int v = 0; v < 8; ++v) {
      t1[rbase + v][cidx]      = (__bf16)silu_f(a0[v] + bv);
      t1[16 + rbase + v][cidx] = (__bf16)silu_f(a1[v] + bv);
    }
  }
  __syncthreads();

  // ---- GEMM2: ef = SiLU([32,128] @ [128,128] + b); aggregate to nodes ----
  {
    v8f a0 = {}, a1 = {};
    gemm_tile2(&t1[0][0], 128, peW2, 4, wave, lane, a0, a1);
    const float bv = eb2[cidx];
    float ef0[8], ef1[8];
#pragma unroll
    for (int v = 0; v < 8; ++v) {
      ef0[v] = silu_f(a0[v] + bv);
      ef1[v] = silu_f(a1[v] + bv);
      t2[rbase + v][cidx]      = (__bf16)ef0[v];
      t2[16 + rbase + v][cidx] = (__bf16)ef1[v];
    }
#pragma unroll
    for (int v = 0; v < 8; ++v) {
      atomic_add_f32(&agg_feat[ncol[rbase + v] * HDIM + cidx], ef0[v]);
      atomic_add_f32(&agg_feat[ncol[16 + rbase + v] * HDIM + cidx], ef1[v]);
    }
  }
  __syncthreads();

  // ---- GEMM3: cf = SiLU(ef @ cW1 + b) ----
  {
    v8f a0 = {}, a1 = {};
    gemm_tile2(&t2[0][0], 128, pcW1, 4, wave, lane, a0, a1);
    const float bv = cb1[cidx];
#pragma unroll
    for (int v = 0; v < 8; ++v) {
      cf[rbase + v][cidx]      = silu_f(a0[v] + bv);
      cf[16 + rbase + v][cidx] = silu_f(a1[v] + bv);
    }
  }
  __syncthreads();

  // ---- cw = cf @ cWc (128-dot, fp32); coord aggregation ----
  if (t < 32) {
    float s = 0.f;
    for (int k = 0; k < HDIM; ++k) s += cf[t][k] * cWc[k];
    const int node = ncol[t];
    atomic_add_f32(&agg_coord[node * 3 + 0], cdiff[t][0] * s);
    atomic_add_f32(&agg_coord[node * 3 + 1], cdiff[t][1] * s);
    atomic_add_f32(&agg_coord[node * 3 + 2], cdiff[t][2] * s);
  }
}

// ---------------------------------------------------------------------------
// Node kernel: upd = SiLU(cat(h,agg) @ nW1 + b) @ nW2 + b; h += upd;
// coords += agg_coord / max(counts,1).   N % 16 == 0.
// ---------------------------------------------------------------------------
__global__ __launch_bounds__(256) void node_kernel(
    float* __restrict__ hcur, float* __restrict__ coords,
    const float* __restrict__ agg_feat, const float* __restrict__ agg_coord,
    const float* __restrict__ counts,
    const __bf16* __restrict__ pnW1, const __bf16* __restrict__ pnW2,
    const float* __restrict__ nb1, const float* __restrict__ nb2) {
  __shared__ __bf16 xt[16][256];
  __shared__ __bf16 t1[16][128];
  const int t = threadIdx.x, wave = t >> 5, lane = t & 31;
  const int n0 = blockIdx.x * 16;
  {
    const int s = t >> 4, j = t & 15, nd = n0 + s;
#pragma unroll
    for (int i = 0; i < 8; ++i) xt[s][j * 8 + i] = (__bf16)hcur[nd * HDIM + j * 8 + i];
#pragma unroll
    for (int i = 0; i < 8; ++i) xt[s][HDIM + j * 8 + i] = (__bf16)agg_feat[nd * HDIM + j * 8 + i];
  }
  __syncthreads();
  const int cidx = wave * 16 + (lane & 15);
  const int rbase = (lane >= 16) ? 8 : 0;
  {
    v8f acc = {};
    acc = gemm_tile(&xt[0][0], 256, pnW1, 8, wave, lane, acc);
    const float bv = nb1[cidx];
#pragma unroll
    for (int v = 0; v < 8; ++v) t1[rbase + v][cidx] = (__bf16)silu_f(acc[v] + bv);
  }
  __syncthreads();
  {
    v8f acc = {};
    acc = gemm_tile(&t1[0][0], 128, pnW2, 4, wave, lane, acc);
    const float bv = nb2[cidx];
#pragma unroll
    for (int v = 0; v < 8; ++v)
      hcur[(n0 + rbase + v) * HDIM + cidx] += acc[v] + bv;   // residual
  }
  if (t < 48) {
    const int n = n0 + t / 3, d = t - (t / 3) * 3;
    coords[n * 3 + d] += agg_coord[n * 3 + d] / fmaxf(counts[n], 1.f);
  }
}

// ---------------------------------------------------------------------------
// Embed-out + coords copy into d_out (tuple: h[N,128] then coords[N,3]).
// ---------------------------------------------------------------------------
__global__ __launch_bounds__(256) void final_kernel(
    const float* __restrict__ hcur, const float* __restrict__ coords,
    const __bf16* __restrict__ pW, const float* __restrict__ bias,
    float* __restrict__ out_h, float* __restrict__ out_c) {
  __shared__ __bf16 xt[16][128];
  const int t = threadIdx.x, wave = t >> 5, lane = t & 31;
  const int n0 = blockIdx.x * 16;
  {
    const int s = t >> 4, j = t & 15, nd = n0 + s;
#pragma unroll
    for (int i = 0; i < 8; ++i) xt[s][j * 8 + i] = (__bf16)hcur[nd * HDIM + j * 8 + i];
  }
  __syncthreads();
  const int cidx = wave * 16 + (lane & 15);
  const int rbase = (lane >= 16) ? 8 : 0;
  v8f acc = {};
  acc = gemm_tile(&xt[0][0], 128, pW, 4, wave, lane, acc);
  const float bv = bias[cidx];
#pragma unroll
  for (int v = 0; v < 8; ++v) out_h[(n0 + rbase + v) * HDIM + cidx] = acc[v] + bv;
  if (t < 48) {
    const int n = n0 + t / 3, d = t - (t / 3) * 3;
    out_c[n * 3 + d] = coords[n * 3 + d];
  }
}

// ---------------------------------------------------------------------------
extern "C" void kernel_launch(void* const* d_in, const int* in_sizes, int n_in,
                              void* d_out, int out_size, void* d_ws, size_t ws_size,
                              hipStream_t stream) {
  const float* h_in      = (const float*)d_in[0];
  const float* coords_in = (const float*)d_in[1];
  const float* eattr     = (const float*)d_in[2];
  const float* emb_in_W  = (const float*)d_in[3];
  const float* emb_in_b  = (const float*)d_in[4];
  const float* emb_out_W = (const float*)d_in[5];
  const float* emb_out_b = (const float*)d_in[6];
  const float* eW1 = (const float*)d_in[7];
  const float* eb1 = (const float*)d_in[8];
  const float* eW2 = (const float*)d_in[9];
  const float* eb2 = (const float*)d_in[10];
  const float* cW1 = (const float*)d_in[11];
  const float* cb1 = (const float*)d_in[12];
  const float* cWc = (const float*)d_in[13];
  const float* nW1 = (const float*)d_in[14];
  const float* nb1 = (const float*)d_in[15];
  const float* nW2 = (const float*)d_in[16];
  const float* nb2 = (const float*)d_in[17];
  const int*   ei  = (const int*)d_in[18];

  char* ws = (char*)d_ws;
  float*  hcur     = (float*)(ws + 0);
  float*  agg_feat = (float*)(ws + 25600000);   // contiguous with agg_coord
  float*  agg_cd   = (float*)(ws + 51200000);
  float*  coords_c = (float*)(ws + 51800000);
  float*  counts   = (float*)(ws + 52400000);
  __bf16* packed   = (__bf16*)(ws + 52600000);

  // Packed fragment offsets (in bf16 elements); 4096 elems per 32-wide k-tile.
  const int PK_EMB_IN  = 0;                         // K=64  -> 2 tiles
  const int PK_EMB_OUT = 8192;                      // K=128 -> 4 tiles
  const int PK_LAYER0  = 24576;
  const int L_STRIDE   = 118784;                    // 9+4+4+8+4 tiles
  const int O_EW1 = 0, O_EW2 = 36864, O_CW1 = 53248, O_NW1 = 69632, O_NW2 = 102400;

  auto pack = [&](const float* src, __bf16* dst, int K) {
    const int total = ((K + 31) / 32) * 4096;
    pack_weight<<<(total + 255) / 256, 256, 0, stream>>>(src, dst, K, total);
  };

  // ---- pack weights (deterministic, re-done every call) ----
  pack(emb_in_W,  packed + PK_EMB_IN,  IN_CH);
  pack(emb_out_W, packed + PK_EMB_OUT, HDIM);
  for (int l = 0; l < NLAYERS; ++l) {
    __bf16* lb = packed + PK_LAYER0 + l * L_STRIDE;
    pack(eW1 + l * 258 * HDIM, lb + O_EW1, 258);
    pack(eW2 + l * HDIM * HDIM, lb + O_EW2, HDIM);
    pack(cW1 + l * HDIM * HDIM, lb + O_CW1, HDIM);
    pack(nW1 + l * 2 * HDIM * HDIM, lb + O_NW1, 2 * HDIM);
    pack(nW2 + l * HDIM * HDIM, lb + O_NW2, HDIM);
  }

  // ---- counts (incoming edges per target), coords working copy ----
  fill_zero<<<(N_NODES + 255) / 256, 256, 0, stream>>>(counts, N_NODES);
  count_kernel<<<N_EDGES / 256, 256, 0, stream>>>(ei, counts);
  hipMemcpyAsync(coords_c, coords_in, N_NODES * 3 * sizeof(float),
                 hipMemcpyDeviceToDevice, stream);

  // ---- embed in ----
  embed_kernel<<<N_NODES / 16, 256, 0, stream>>>(h_in, packed + PK_EMB_IN,
                                                 emb_in_b, hcur);

  // ---- layers ----
  const int agg_elems = N_NODES * HDIM + N_NODES * 3;  // agg_feat + agg_coord
  for (int l = 0; l < NLAYERS; ++l) {
    __bf16* lb = packed + PK_LAYER0 + l * L_STRIDE;
    fill_zero<<<(agg_elems + 255) / 256, 256, 0, stream>>>(agg_feat, agg_elems);
    edge_kernel<<<N_EDGES / 32, 256, 0, stream>>>(
        ei, coords_c, hcur, eattr,
        lb + O_EW1, lb + O_EW2, lb + O_CW1,
        eb1 + l * HDIM, eb2 + l * HDIM, cb1 + l * HDIM, cWc + l * HDIM,
        agg_feat, agg_cd);
    node_kernel<<<N_NODES / 16, 256, 0, stream>>>(
        hcur, coords_c, agg_feat, agg_cd, counts,
        lb + O_NW1, lb + O_NW2, nb1 + l * HDIM, nb2 + l * HDIM);
  }

  // ---- embed out + coords copy into d_out ----
  float* out_h = (float*)d_out;
  float* out_c = out_h + (size_t)N_NODES * HDIM;
  final_kernel<<<N_NODES / 16, 256, 0, stream>>>(hcur, coords_c,
                                                 packed + PK_EMB_OUT, emb_out_b,
                                                 out_h, out_c);
}
